// OldDuelingModel_62053687493050
// MI455X (gfx1250) — matrix operand order repaired and checked by reading, not compiled, not run on hardware
//
#include <hip/hip_runtime.h>

// ---------------------------------------------------------------------------
// Problem constants (from reference): B=4096, D=4096, H=1024, A=18
// ---------------------------------------------------------------------------
constexpr int BATCH = 4096;
constexpr int DIM   = 4096;
constexpr int HID   = 1024;
constexpr int NACT  = 18;

typedef __attribute__((ext_vector_type(16))) __bf16        v16bf;
typedef __attribute__((ext_vector_type(8)))  float         v8f;
typedef __attribute__((ext_vector_type(4)))  float         f32x4;
typedef __attribute__((ext_vector_type(4)))  unsigned int  u32x4;
typedef __attribute__((ext_vector_type(4)))  int           i32x4;

typedef __attribute__((address_space(1))) void  gas_void;
typedef __attribute__((address_space(3))) void  las_void;
typedef __attribute__((address_space(1))) i32x4 gas_i32x4;
typedef __attribute__((address_space(3))) i32x4 las_i32x4;

// f(x) = sign(x) * sqrt(|x|)
__device__ __forceinline__ float fsig(float x) {
    return copysignf(sqrtf(fabsf(x)), x);
}

// float -> bf16 bits, round-to-nearest-even
__device__ __forceinline__ unsigned short f2bf(float f) {
    unsigned int u = __float_as_uint(f);
    u += 0x7FFFu + ((u >> 16) & 1u);
    return (unsigned short)(u >> 16);
}
__device__ __forceinline__ unsigned int pk2(float a, float b) {
    return (unsigned int)f2bf(a) | ((unsigned int)f2bf(b) << 16);
}

// ---- CDNA5 async global->LDS copy (ASYNCcnt-tracked), 16 bytes per lane ----
__device__ __forceinline__ void async_b128(const void* g, void* l) {
#if __has_builtin(__builtin_amdgcn_global_load_async_to_lds_b128)
    __builtin_amdgcn_global_load_async_to_lds_b128(
        (gas_i32x4*)(gas_void*)(void*)g,
        (las_i32x4*)(las_void*)l, 0, 0);
#else
    unsigned int lofs = (unsigned int)(unsigned long long)(las_void*)l;
    asm volatile("global_load_async_to_lds_b128 %0, %1, off"
                 :: "v"(lofs), "v"(g) : "memory");
#endif
}

#if __has_builtin(__builtin_amdgcn_s_wait_asynccnt)
#define WAIT_ASYNC(n) __builtin_amdgcn_s_wait_asynccnt(n)
#else
#define WAIT_ASYNC(n) asm volatile("s_wait_asynccnt %0" :: "i"(n) : "memory")
#endif

union FragBits {
    u32x4 q[2];
    v16bf v;
};

// A-matrix 16x32 bf16 fragment (ISA 7.12.2): lanes 0-15 hold K=0..7 / K=16..23;
// lanes 16-31 hold K=8..15 / K=24..31. `row` = 32-element contiguous K-row.
__device__ __forceinline__ v16bf load_a_frag(const unsigned short* row, int hi) {
    FragBits u;
    u.q[0] = *reinterpret_cast<const u32x4*>(row + (hi ? 8  : 0));
    u.q[1] = *reinterpret_cast<const u32x4*>(row + (hi ? 24 : 16));
    return u.v;
}

// B-matrix 32x16 bf16 fragment: lanes 0-15 hold K=0..7 / K=16..23; lanes 16-31 K=8..15 / K=24..31.
__device__ __forceinline__ v16bf load_b_frag(const unsigned short* row, int hi) {
    FragBits u;
    u.q[0] = *reinterpret_cast<const u32x4*>(row + hi * 8);
    u.q[1] = *reinterpret_cast<const u32x4*>(row + hi * 8 + 16);
    return u.v;
}

// ---------------------------------------------------------------------------
// Prepass 1: xb = bf16(x), xnb = bf16(x * f(e_in)) — done ONCE (not per N-tile)
// ---------------------------------------------------------------------------
__global__ __launch_bounds__(256)
void prep_x(const float* __restrict__ x, const float* __restrict__ ein,
            unsigned short* __restrict__ xb, unsigned short* __restrict__ xnb)
{
    const size_t base = ((size_t)blockIdx.x * 256 + threadIdx.x) * 8;
    const f32x4 a0 = *reinterpret_cast<const f32x4*>(x + base);
    const f32x4 a1 = *reinterpret_cast<const f32x4*>(x + base + 4);
    const f32x4 e0 = *reinterpret_cast<const f32x4*>(ein + base);
    const f32x4 e1 = *reinterpret_cast<const f32x4*>(ein + base + 4);

    u32x4 ob, on;
    ob.x = pk2(a0.x, a0.y); ob.y = pk2(a0.z, a0.w);
    ob.z = pk2(a1.x, a1.y); ob.w = pk2(a1.z, a1.w);
    on.x = pk2(a0.x * fsig(e0.x), a0.y * fsig(e0.y));
    on.y = pk2(a0.z * fsig(e0.z), a0.w * fsig(e0.w));
    on.z = pk2(a1.x * fsig(e1.x), a1.y * fsig(e1.y));
    on.w = pk2(a1.z * fsig(e1.z), a1.w * fsig(e1.w));
    *reinterpret_cast<u32x4*>(xb  + base) = ob;
    *reinterpret_cast<u32x4*>(xnb + base) = on;
}

// ---------------------------------------------------------------------------
// Prepass 2: wbT[n][k] = bf16(W[k][n]), sbT likewise (64x64 tile transpose)
// ---------------------------------------------------------------------------
__global__ __launch_bounds__(256)
void prep_w(const float* __restrict__ Wh, const float* __restrict__ Sh,
            unsigned short* __restrict__ wbT, unsigned short* __restrict__ sbT)
{
    __shared__ __align__(16) unsigned short tw[64][72];
    __shared__ __align__(16) unsigned short ts[64][72];

    const int t  = threadIdx.x;
    const int k0 = blockIdx.x * 64;   // over DIM
    const int n0 = blockIdx.y * 64;   // over HID

#pragma unroll
    for (int it = 0; it < 4; ++it) {
        const int idx = t + it * 256;        // 0..1023 float4 slots
        const int r   = idx >> 4;            // k within tile
        const int c4  = (idx & 15) * 4;      // n within tile
        const size_t g = (size_t)(k0 + r) * HID + n0 + c4;
        const f32x4 w4 = *reinterpret_cast<const f32x4*>(Wh + g);
        const f32x4 s4 = *reinterpret_cast<const f32x4*>(Sh + g);
        tw[r][c4 + 0] = f2bf(w4.x); tw[r][c4 + 1] = f2bf(w4.y);
        tw[r][c4 + 2] = f2bf(w4.z); tw[r][c4 + 3] = f2bf(w4.w);
        ts[r][c4 + 0] = f2bf(s4.x); ts[r][c4 + 1] = f2bf(s4.y);
        ts[r][c4 + 2] = f2bf(s4.z); ts[r][c4 + 3] = f2bf(s4.w);
    }
    __syncthreads();

#pragma unroll
    for (int it = 0; it < 2; ++it) {
        const int idx = t + it * 256;        // 0..511 chunks of 8 bf16
        const int n   = idx >> 3;
        const int k8  = (idx & 7) * 8;
        u32x4 o, p;
        o.x = (unsigned)tw[k8+0][n] | ((unsigned)tw[k8+1][n] << 16);
        o.y = (unsigned)tw[k8+2][n] | ((unsigned)tw[k8+3][n] << 16);
        o.z = (unsigned)tw[k8+4][n] | ((unsigned)tw[k8+5][n] << 16);
        o.w = (unsigned)tw[k8+6][n] | ((unsigned)tw[k8+7][n] << 16);
        p.x = (unsigned)ts[k8+0][n] | ((unsigned)ts[k8+1][n] << 16);
        p.y = (unsigned)ts[k8+2][n] | ((unsigned)ts[k8+3][n] << 16);
        p.z = (unsigned)ts[k8+4][n] | ((unsigned)ts[k8+5][n] << 16);
        p.w = (unsigned)ts[k8+6][n] | ((unsigned)ts[k8+7][n] << 16);
        const size_t g = (size_t)(n0 + n) * DIM + k0 + k8;
        *reinterpret_cast<u32x4*>(wbT + g) = o;
        *reinterpret_cast<u32x4*>(sbT + g) = p;
    }
}

// ---------------------------------------------------------------------------
// Main GEMM: features = relu(xb@wb + fout*(xnb@sb) + b + bs*fout)
// Block 128x64, K-tile 32, 8 waves (wave tile 32x32, 2 noise paths),
// double-buffered LDS fed by async global->LDS b128 copies.
// ---------------------------------------------------------------------------
#define LDA 40  // LDS K-row stride in bf16 elements (32 + 8 pad)

__global__ __launch_bounds__(256)
void noisy_gemm1(const unsigned short* __restrict__ xb,
                 const unsigned short* __restrict__ xnb,
                 const unsigned short* __restrict__ wbT,
                 const unsigned short* __restrict__ sbT,
                 const float* __restrict__ bh,  const float* __restrict__ bsh,
                 const float* __restrict__ eout, float* __restrict__ feat)
{
    __shared__ __align__(16) unsigned short As[2][128 * LDA];
    __shared__ __align__(16) unsigned short An[2][128 * LDA];
    __shared__ __align__(16) unsigned short Bw[2][64 * LDA];
    __shared__ __align__(16) unsigned short Bs[2][64 * LDA];

    const int t    = threadIdx.x;
    const int lane = t & 31;
    const int wv   = t >> 5;
    const int hi   = lane >> 4;
    const int ar   = lane & 15;

    const int m0 = blockIdx.x * 128;
    const int n0 = blockIdx.y * 64;
    const int rowOff = (wv & 3) * 32;
    const int colOff = (wv >> 2) * 32;

    // Issue one double-buffer stage: 6 async b128 copies per thread.
    auto stage = [&](int s, int k0) {
#pragma unroll
        for (int it = 0; it < 2; ++it) {
            const int o  = t + it * 256;        // 0..511
            const int r  = o >> 2;              // A row (0..127)
            const int ch = (o & 3) * 8;         // 8 bf16 = 16B chunk
            const size_t g = (size_t)(m0 + r) * DIM + k0 + ch;
            async_b128(xb  + g, &As[s][r * LDA + ch]);
            async_b128(xnb + g, &An[s][r * LDA + ch]);
        }
        {
            const int r  = t >> 2;              // B row n (0..63)
            const int ch = (t & 3) * 8;
            const size_t g = (size_t)(n0 + r) * DIM + k0 + ch;
            async_b128(wbT + g, &Bw[s][r * LDA + ch]);
            async_b128(sbT + g, &Bs[s][r * LDA + ch]);
        }
    };

    v8f accw[2][2], accs[2][2];
    const v8f vzero = {0.f, 0.f, 0.f, 0.f, 0.f, 0.f, 0.f, 0.f};
#pragma unroll
    for (int i = 0; i < 2; ++i)
#pragma unroll
        for (int j = 0; j < 2; ++j) { accw[i][j] = vzero; accs[i][j] = vzero; }

    constexpr int NK = DIM / 32;
    stage(0, 0);

    for (int kt = 0; kt < NK; ++kt) {
        const int s = kt & 1;
        if (kt + 1 < NK) {
            stage(s ^ 1, (kt + 1) * 32);   // prefetch next tile into other buffer
            WAIT_ASYNC(6);                 // current tile's 6 copies complete
        } else {
            WAIT_ASYNC(0);
        }
        __syncthreads();                   // all waves' copies visible

        v16bf ax[2], an[2], bw[2], bs2[2];
#pragma unroll
        for (int i = 0; i < 2; ++i) {
            ax[i] = load_a_frag(&As[s][(rowOff + i * 16 + ar) * LDA], hi);
            an[i] = load_a_frag(&An[s][(rowOff + i * 16 + ar) * LDA], hi);
        }
#pragma unroll
        for (int j = 0; j < 2; ++j) {
            bw[j]  = load_b_frag(&Bw[s][(colOff + j * 16 + ar) * LDA], hi);
            bs2[j] = load_b_frag(&Bs[s][(colOff + j * 16 + ar) * LDA], hi);
        }
#pragma unroll
        for (int i = 0; i < 2; ++i)
#pragma unroll
            for (int j = 0; j < 2; ++j) {
                accw[i][j] = __builtin_amdgcn_wmma_f32_16x16x32_bf16(
                    false, ax[i], false, bw[j], (short)0, accw[i][j], false, false);
                accs[i][j] = __builtin_amdgcn_wmma_f32_16x16x32_bf16(
                    false, an[i], false, bs2[j], (short)0, accs[i][j], false, false);
            }
        __syncthreads();                   // protect buffer before next overwrite
    }

    // Epilogue: out = relu(accW + fout*accS + b + bs*fout)
    const int mhi = hi * 8;
#pragma unroll
    for (int i = 0; i < 2; ++i)
#pragma unroll
        for (int j = 0; j < 2; ++j)
#pragma unroll
            for (int r = 0; r < 8; ++r) {
                const int row = m0 + rowOff + i * 16 + mhi + r;
                const int col = n0 + colOff + j * 16 + ar;
                const float fo = fsig(eout[(size_t)row * HID + col]);
                float v = accw[i][j][r] + fo * accs[i][j][r]
                        + bh[col] + bsh[col] * fo;
                feat[(size_t)row * HID + col] = v > 0.f ? v : 0.f;
            }
}

// ---------------------------------------------------------------------------
// Heads: dueling value/advantage + mean-subtract. One wave per batch row.
// ---------------------------------------------------------------------------
__global__ __launch_bounds__(256)
void heads_kernel(const float* __restrict__ feat,
                  const float* __restrict__ Wa, const float* __restrict__ Sa,
                  const float* __restrict__ ba, const float* __restrict__ bsa,
                  const float* __restrict__ Wv, const float* __restrict__ Sv,
                  const float* __restrict__ bv, const float* __restrict__ bsv,
                  const float* __restrict__ eina, const float* __restrict__ eouta,
                  const float* __restrict__ einv, const float* __restrict__ eoutv,
                  float* __restrict__ out)
{
    __shared__ float sx[8][512];
    __shared__ float sn[8][512];
    __shared__ float sadv[8][20];

    const int t    = threadIdx.x;
    const int lane = t & 31;
    const int wv   = t >> 5;
    const int b    = blockIdx.x * 8 + wv;

    const float* fr = feat + (size_t)b * HID;

    float accv = 0.f, accsv = 0.f;
    for (int i = lane; i < 512; i += 32) {
        const float xa = fr[i];
        sx[wv][i] = xa;
        sn[wv][i] = xa * fsig(eina[(size_t)b * 512 + i]);
        const float xv = fr[512 + i];
        accv  += xv * Wv[i];
        accsv += xv * fsig(einv[(size_t)b * 512 + i]) * Sv[i];
    }
#pragma unroll
    for (int off = 16; off > 0; off >>= 1) {
        accv  += __shfl_xor(accv,  off, 32);
        accsv += __shfl_xor(accsv, off, 32);
    }
    __syncthreads();

    if (lane < NACT) {
        float a = 0.f, s = 0.f;
        for (int k = 0; k < 512; ++k) {
            a += sx[wv][k] * Wa[k * NACT + lane];
            s += sn[wv][k] * Sa[k * NACT + lane];
        }
        const float fo = fsig(eouta[(size_t)b * NACT + lane]);
        sadv[wv][lane] = a + fo * s + ba[lane] + bsa[lane] * fo;
    }
    __syncthreads();

    if (lane < NACT) {
        float m = 0.f;
#pragma unroll
        for (int o = 0; o < NACT; ++o) m += sadv[wv][o];
        m *= (1.0f / (float)NACT);
        const float fov   = fsig(eoutv[b]);
        const float value = accv + fov * accsv + bv[0] + bsv[0] * fov;
        out[(size_t)b * NACT + lane] = value + sadv[wv][lane] - m;
    }
}

// ---------------------------------------------------------------------------
// Launcher. setup_inputs() order:
//  0:x 1:W_h 2:S_h 3:b_h 4:bs_h 5:W_a 6:S_a 7:b_a 8:bs_a 9:W_v 10:S_v 11:b_v
//  12:bs_v 13:e_in_h 14:e_out_h 15:e_in_a 16:e_out_a 17:e_in_v 18:e_out_v
// ---------------------------------------------------------------------------
extern "C" void kernel_launch(void* const* d_in, const int* in_sizes, int n_in,
                              void* d_out, int out_size, void* d_ws, size_t ws_size,
                              hipStream_t stream) {
    const float* x     = (const float*)d_in[0];
    const float* Wh    = (const float*)d_in[1];
    const float* Sh    = (const float*)d_in[2];
    const float* bh    = (const float*)d_in[3];
    const float* bsh   = (const float*)d_in[4];
    const float* Wa    = (const float*)d_in[5];
    const float* Sa    = (const float*)d_in[6];
    const float* ba    = (const float*)d_in[7];
    const float* bsa   = (const float*)d_in[8];
    const float* Wv    = (const float*)d_in[9];
    const float* Sv    = (const float*)d_in[10];
    const float* bv    = (const float*)d_in[11];
    const float* bsv   = (const float*)d_in[12];
    const float* einh  = (const float*)d_in[13];
    const float* eouth = (const float*)d_in[14];
    const float* eina  = (const float*)d_in[15];
    const float* eouta = (const float*)d_in[16];
    const float* einv  = (const float*)d_in[17];
    const float* eoutv = (const float*)d_in[18];

    // Workspace layout (bytes):
    //   feat : 16 MB  | xb : 32 MB | xnb : 32 MB | wbT : 8 MB | sbT : 8 MB
    char* ws = (char*)d_ws;
    float*          feat = (float*)ws;
    unsigned short* xb   = (unsigned short*)(ws + (16u << 20));
    unsigned short* xnb  = (unsigned short*)(ws + (48u << 20));
    unsigned short* wbT  = (unsigned short*)(ws + (80u << 20));
    unsigned short* sbT  = (unsigned short*)(ws + (88u << 20));
    float* outp = (float*)d_out;

    prep_x<<<(size_t)BATCH * DIM / (256 * 8), 256, 0, stream>>>(x, einh, xb, xnb);
    prep_w<<<dim3(DIM / 64, HID / 64), 256, 0, stream>>>(Wh, Sh, wbT, sbT);

    noisy_gemm1<<<dim3(BATCH / 128, HID / 64), 256, 0, stream>>>(
        xb, xnb, wbT, sbT, bh, bsh, eouth, feat);

    heads_kernel<<<BATCH / 8, 256, 0, stream>>>(feat, Wa, Sa, ba, bsa,
                                                Wv, Sv, bv, bsv,
                                                eina, eouta, einv, eoutv, outp);
}
